// LSGAttentionProduct_6485400617276
// MI455X (gfx1250) — matrix operand order repaired and checked by reading, not compiled, not run on hardware
//
#include <hip/hip_runtime.h>
#include <hip/hip_bf16.h>
#include <float.h>

typedef _Float16 half_t;
typedef __attribute__((ext_vector_type(16))) _Float16 v16h;
typedef __attribute__((ext_vector_type(8)))  _Float16 v8h;
typedef __attribute__((ext_vector_type(2)))  __fp16   v2pk;   // return type of cvt_pkrtz
typedef __attribute__((ext_vector_type(8)))  float    v8f;
typedef __attribute__((ext_vector_type(4)))  float    v4f;

#define N_B   2
#define H_B   12
#define T_B   4096
#define D_B   64
#define G_B   64
#define TS_B  1024
#define BLK   128
#define KTOT  704      // 64 global + 256 sparse + 384 local
#define NCHUNK 22      // 704 / 32
#define NEGV  (-3.402823466e38f)

#define WMMA_F16(A, B, C) \
  __builtin_amdgcn_wmma_f32_16x16x32_f16(false, (A), false, (B), (short)0, (C), false, false)

// A-fragment (16x32 f16) from row-major LDS [rows][ldk]:
// lanes 0-15: row=lane, K halves {0..7, 16..23}; lanes 16-31: row=lane-16, K {8..15, 24..31}
__device__ __forceinline__ v16h load_afrag(const half_t* base, int ldk, int row, int kstart) {
  int lane = threadIdx.x & 31;
  int kb   = (lane & 16) >> 1;                       // 0 or 8
  const half_t* p = base + (size_t)(row + (lane & 15)) * ldk + kstart + kb;
  v8h lo = *(const v8h*)(p);                          // K = kb .. kb+7
  v8h hi = *(const v8h*)(p + 16);                     // K = kb+16 .. kb+23
  v16h r;
#pragma unroll
  for (int i = 0; i < 8; ++i) { r[i] = lo[i]; r[i + 8] = hi[i]; }
  return r;
}

// B-fragment (32x16 f16), element (k, n) = base[(nrow + n)*ldk + kstart + k]:
// lanes 0-15: n=lane, K {0..7, 16..23}; lanes 16-31: n=lane-16, K {8..15, 24..31}
__device__ __forceinline__ v16h load_bfrag(const half_t* base, int ldk, int nrow, int kstart) {
  int lane = threadIdx.x & 31;
  const half_t* p = base + (size_t)(nrow + (lane & 15)) * ldk + kstart + ((lane & 16) >> 1);
  v8h lo = *(const v8h*)(p);
  v8h hi = *(const v8h*)(p + 16);
  v16h r;
#pragma unroll
  for (int i = 0; i < 8; ++i) { r[i] = lo[i]; r[i + 8] = hi[i]; }
  return r;
}

// max-reduction across the 16-lane half group: single ds_swizzle per step (group-of-32 xor)
__device__ __forceinline__ float max_red16(float x) {
  x = fmaxf(x, __int_as_float(__builtin_amdgcn_ds_swizzle(__float_as_int(x), 0x041f)));
  x = fmaxf(x, __int_as_float(__builtin_amdgcn_ds_swizzle(__float_as_int(x), 0x081f)));
  x = fmaxf(x, __int_as_float(__builtin_amdgcn_ds_swizzle(__float_as_int(x), 0x101f)));
  x = fmaxf(x, __int_as_float(__builtin_amdgcn_ds_swizzle(__float_as_int(x), 0x201f)));
  return x;
}

// Gather one key/value row pointer + its additive mask for global key index kk in [0, 704)
__device__ __forceinline__ const float* gather_row(
    int kk, int b,
    const float* layNH, const float* slayNH, const float* glayNH,
    const float* amN, const float* smN, const float* gmN, float* mval)
{
  if (kk < 64) {                                     // global segment
    *mval = gmN[kk];
    return glayNH + (size_t)kk * D_B;
  }
  if (kk < 320) {                                    // sparse segment (2x128)
    int s   = kk - 64;
    int tok = (s < 128) ? (b * 32 - 160 + s) : (b * 32 + 64 + (s - 128));
    if (tok >= 0 && tok < TS_B) { *mval = smN[tok]; return slayNH + (size_t)tok * D_B; }
    *mval = NEGV; return nullptr;
  }
  int tok = (b - 1) * BLK + (kk - 320);              // local segment (3x128)
  if (tok >= 0 && tok < T_B) { *mval = amN[tok]; return layNH + (size_t)tok * D_B; }
  *mval = NEGV; return nullptr;
}

// Issue the global gathers for chunk c into registers.
// tid < 128  : K loader, thread owns (row = tid/4, 16 dims at seg*16) -> regv[0..15]
// tid >= 128 : V loader, thread owns (row pair rp, 8 dims at cg*8) -> regv[0..7]=row rp, [8..15]=row rp+1
__device__ __forceinline__ void fetch_chunk(
    int c, int b, int tid,
    const float* kNH, const float* skNH, const float* gkNH,
    const float* vNH, const float* svNH, const float* gvNH,
    const float* amN, const float* smN, const float* gmN,
    float* regv, float* mval)
{
  if (tid < 128) {
    int r = tid >> 2, seg = tid & 3;
    const float* src = gather_row(c * 32 + r, b, kNH, skNH, gkNH, amN, smN, gmN, mval);
    if (src) {
      const float* s16 = src + seg * 16;
#pragma unroll
      for (int p4 = 0; p4 < 4; ++p4) {
        v4f x = *(const v4f*)(s16 + p4 * 4);
#pragma unroll
        for (int j = 0; j < 4; ++j) regv[p4 * 4 + j] = x[j];
      }
    } else {
#pragma unroll
      for (int j = 0; j < 16; ++j) regv[j] = 0.0f;
    }
  } else {
    int t2 = tid - 128;
    int cg = t2 & 7;            // dim group (8 dims)
    int rp = (t2 >> 3) * 2;     // row pair
    float dummy;
    const float* s0 = gather_row(c * 32 + rp,     b, vNH, svNH, gvNH, amN, smN, gmN, &dummy);
    const float* s1 = gather_row(c * 32 + rp + 1, b, vNH, svNH, gvNH, amN, smN, gmN, &dummy);
    if (s0) {
      v4f x0 = *(const v4f*)(s0 + cg * 8);
      v4f x1 = *(const v4f*)(s0 + cg * 8 + 4);
#pragma unroll
      for (int j = 0; j < 4; ++j) { regv[j] = x0[j]; regv[4 + j] = x1[j]; }
    } else {
#pragma unroll
      for (int j = 0; j < 8; ++j) regv[j] = 0.0f;
    }
    if (s1) {
      v4f x0 = *(const v4f*)(s1 + cg * 8);
      v4f x1 = *(const v4f*)(s1 + cg * 8 + 4);
#pragma unroll
      for (int j = 0; j < 4; ++j) { regv[8 + j] = x0[j]; regv[12 + j] = x1[j]; }
    } else {
#pragma unroll
      for (int j = 0; j < 8; ++j) regv[8 + j] = 0.0f;
    }
    *mval = 0.0f;
  }
}

// Convert + store registers into one K/V LDS buffer (packed f16 pair stores).
__device__ __forceinline__ void store_chunk(
    int tid, half_t* sKbuf, half_t* sVTbuf, float* sMaskbuf,
    const float* regv, float mval)
{
  if (tid < 128) {
    int r = tid >> 2, seg = tid & 3;
    half_t* dst = sKbuf + r * D_B + seg * 16;
#pragma unroll
    for (int j2 = 0; j2 < 8; ++j2) {
      v2pk pk = __builtin_amdgcn_cvt_pkrtz(regv[2 * j2], regv[2 * j2 + 1]);
      *(v2pk*)(dst + 2 * j2) = pk;
    }
    if (seg == 0) sMaskbuf[r] = mval;
  } else {
    int t2 = tid - 128;
    int cg = t2 & 7;
    int rp = (t2 >> 3) * 2;
#pragma unroll
    for (int j = 0; j < 8; ++j) {
      int col = cg * 8 + j;
      v2pk pk = __builtin_amdgcn_cvt_pkrtz(regv[j], regv[8 + j]);  // rows rp, rp+1
      *(v2pk*)(sVTbuf + col * 32 + rp) = pk;
    }
  }
}

__global__ __launch_bounds__(256) void lsg_attn_kernel(
    const float* __restrict__ q,  const float* __restrict__ k,  const float* __restrict__ v,
    const float* __restrict__ am, const float* __restrict__ sk, const float* __restrict__ sv,
    const float* __restrict__ sm, const float* __restrict__ gk, const float* __restrict__ gv,
    const float* __restrict__ gm, float* __restrict__ out)
{
  __shared__ __align__(128) half_t sQ[BLK * D_B];         // 16 KB, Q tile f16
  __shared__ __align__(128) half_t sK[2 * 32 * D_B];      // 8 KB, K chunks [buf][key][dim]
  __shared__ __align__(128) half_t sVT[2 * D_B * 32];     // 8 KB, V chunks transposed [buf][dim][key]
  __shared__ __align__(128) half_t sP[8 * 16 * 32];       // 8 KB, per-wave P scratch
  __shared__ float sMask[2 * 32];

  const int b    = blockIdx.x;
  const int hIdx = blockIdx.y;
  const int nIdx = blockIdx.z;
  const int tid  = threadIdx.x;
  const int lane = tid & 31;
  const int wave = tid >> 5;

  const size_t nh = (size_t)(nIdx * H_B + hIdx);
  const float* qNH  = q  + nh * T_B  * D_B;
  const float* kNH  = k  + nh * T_B  * D_B;
  const float* vNH  = v  + nh * T_B  * D_B;
  const float* skNH = sk + nh * TS_B * D_B;
  const float* svNH = sv + nh * TS_B * D_B;
  const float* gkNH = gk + nh * G_B  * D_B;
  const float* gvNH = gv + nh * G_B  * D_B;
  const float* amN  = am + (size_t)nIdx * T_B;
  const float* smN  = sm + (size_t)nIdx * TS_B;
  const float* gmN  = gm + (size_t)nIdx * G_B;

  // ---- stage Q tile (128x64 fp32 -> f16 LDS, packed converts) ----
  for (int t = tid; t < 512; t += 256) {
    int r = t >> 2, seg = t & 3;
    const float* src = qNH + (size_t)(b * BLK + r) * D_B + seg * 16;
    half_t* dst = sQ + r * D_B + seg * 16;
#pragma unroll
    for (int p4 = 0; p4 < 4; ++p4) {
      v4f x = *(const v4f*)(src + p4 * 4);
      *(v2pk*)(dst + p4 * 4)     = __builtin_amdgcn_cvt_pkrtz(x[0], x[1]);
      *(v2pk*)(dst + p4 * 4 + 2) = __builtin_amdgcn_cvt_pkrtz(x[2], x[3]);
    }
  }

  // ---- prologue: fetch + stage chunk 0 into buffer 0 ----
  {
    float regv[16]; float mval;
    fetch_chunk(0, b, tid, kNH, skNH, gkNH, vNH, svNH, gvNH, amN, smN, gmN, regv, &mval);
    store_chunk(tid, sK, sVT, sMask, regv, mval);
  }
  __syncthreads();

  // per-wave Q A-fragments (registers for whole kernel)
  const v16h qa0 = load_afrag(sQ, D_B, wave * 16, 0);
  const v16h qa1 = load_afrag(sQ, D_B, wave * 16, 32);

  // constant ones B-fragment (for WMMA row-sum)
  v16h ones;
#pragma unroll
  for (int i = 0; i < 16; ++i) ones[i] = (half_t)1.0f;

  // flash-attention state in C-fragment layout
  v8f o0 = {}, o1 = {}, o2 = {}, o3 = {};
  float mrow[8], lrow[8];
#pragma unroll
  for (int vv = 0; vv < 8; ++vv) { mrow[vv] = NEGV; lrow[vv] = 0.0f; }

  const int rowoff = (lane & 16) >> 1;   // 0 or 8
  const int colb   = lane & 15;
  half_t* myP = sP + wave * (16 * 32);

  for (int c = 0; c < NCHUNK; ++c) {
    const int cur = c & 1;
    const int nxt = cur ^ 1;
    half_t* cK  = sK  + cur * (32 * D_B);
    half_t* cVT = sVT + cur * (D_B * 32);
    float*  cM  = sMask + cur * 32;

    // issue next chunk's global gathers NOW (overlap with WMMA below)
    float regv[16]; float mval = 0.0f;
    const bool have_next = (c + 1 < NCHUNK);
    if (have_next)
      fetch_chunk(c + 1, b, tid, kNH, skNH, gkNH, vNH, svNH, gvNH, amN, smN, gmN, regv, &mval);

    // ---- S = Q . K^T (16x32 tile per wave, f32 accum) ----
    v8f s0 = {}, s1 = {};
    s0 = WMMA_F16(qa0, load_bfrag(cK, D_B,  0,  0), s0);
    s0 = WMMA_F16(qa1, load_bfrag(cK, D_B,  0, 32), s0);
    s1 = WMMA_F16(qa0, load_bfrag(cK, D_B, 16,  0), s1);
    s1 = WMMA_F16(qa1, load_bfrag(cK, D_B, 16, 32), s1);

    const float mk0 = cM[colb];
    const float mk1 = cM[colb + 16];

    // ---- online softmax: per-row max via ds_swizzle butterfly ----
    float al[8];
#pragma unroll
    for (int vv = 0; vv < 8; ++vv) {
      float a  = s0[vv] * 0.125f + mk0;              // 1/sqrt(64)
      float bb = s1[vv] * 0.125f + mk1;
      float mx = max_red16(fmaxf(a, bb));
      float mn = fmaxf(mrow[vv], mx);
      float alpha = __expf(mrow[vv] - mn);
      mrow[vv] = mn;
      al[vv] = alpha;
      float p0 = __expf(a - mn);
      float p1 = __expf(bb - mn);
      myP[(vv + rowoff) * 32 + colb]      = (half_t)p0;
      myP[(vv + rowoff) * 32 + 16 + colb] = (half_t)p1;
    }
#pragma unroll
    for (int vv = 0; vv < 8; ++vv) {
      o0[vv] *= al[vv]; o1[vv] *= al[vv]; o2[vv] *= al[vv]; o3[vv] *= al[vv];
    }

    // ---- P fragment; row-sum via WMMA with ones; O += P . V ----
    const v16h pa = load_afrag(myP, 32, 0, 0);
    v8f rs = {};
    rs = WMMA_F16(pa, ones, rs);                     // rs[vv] = rowsum(P row vv+rowoff)
#pragma unroll
    for (int vv = 0; vv < 8; ++vv) lrow[vv] = lrow[vv] * al[vv] + rs[vv];

    o0 = WMMA_F16(pa, load_bfrag(cVT, 32,  0, 0), o0);
    o1 = WMMA_F16(pa, load_bfrag(cVT, 32, 16, 0), o1);
    o2 = WMMA_F16(pa, load_bfrag(cVT, 32, 32, 0), o2);
    o3 = WMMA_F16(pa, load_bfrag(cVT, 32, 48, 0), o3);

    // ---- stage next chunk into the other buffer, then one barrier ----
    if (have_next)
      store_chunk(tid, sK + nxt * (32 * D_B), sVT + nxt * (D_B * 32), sMask + nxt * 32,
                  regv, mval);
    __syncthreads();
  }

  // ---- normalize and store (fp32 output) ----
  float* outNH = out + nh * T_B * D_B + (size_t)(b * BLK + wave * 16) * D_B;
#pragma unroll
  for (int vv = 0; vv < 8; ++vv) {
    float inv = 1.0f / lrow[vv];
    float* orow = outNH + (size_t)(vv + rowoff) * D_B;
    orow[colb]      = o0[vv] * inv;
    orow[16 + colb] = o1[vv] * inv;
    orow[32 + colb] = o2[vv] * inv;
    orow[48 + colb] = o3[vv] * inv;
  }
}

extern "C" void kernel_launch(void* const* d_in, const int* in_sizes, int n_in,
                              void* d_out, int out_size, void* d_ws, size_t ws_size,
                              hipStream_t stream) {
  (void)in_sizes; (void)n_in; (void)out_size; (void)d_ws; (void)ws_size;
  const float* q  = (const float*)d_in[0];
  const float* k  = (const float*)d_in[1];
  const float* v  = (const float*)d_in[2];
  const float* am = (const float*)d_in[3];
  const float* sk = (const float*)d_in[4];
  const float* sv = (const float*)d_in[5];
  const float* sm = (const float*)d_in[6];
  const float* gk = (const float*)d_in[7];
  const float* gv = (const float*)d_in[8];
  const float* gm = (const float*)d_in[9];
  float* out = (float*)d_out;

  dim3 grid(T_B / BLK, H_B, N_B);   // (32, 12, 2)
  dim3 block(256);                  // 8 wave32 waves, one 16-row Q strip each
  lsg_attn_kernel<<<grid, block, 0, stream>>>(q, k, v, am, sk, sv, sm, gk, gv, gm, out);
}
